// funcGNN_23931557773531
// MI455X (gfx1250) — matrix operand order repaired and checked by reading, not compiled, not run on hardware
//
#include <hip/hip_runtime.h>
#include <hip/hip_bf16.h>
#include <math.h>

// ---------------------------------------------------------------------------
// funcGNN / SimGNN forward for MI455X (gfx1250, wave32, WMMA)
//
//   (a) edge scatter-mean  : float4 gather + L2-resident float atomics
//   (b) SAGE linear layers : v_wmma_f32_16x16x32_f16, fully unrolled K,
//                            2 M-tiles per wave (B-fragment reuse x2),
//                            pre-packed f16 B fragments (2x b128 each),
//                            fused bias + root GEMM + L2 normalize (+ReLU)
//   (c) attention pooling  : wave reductions + LDS-combined atomics
//   (d) MLP head           : single wave
// ---------------------------------------------------------------------------

typedef __attribute__((ext_vector_type(16))) _Float16 v16h;
typedef __attribute__((ext_vector_type(8)))  float    v8f;

#define N_NODES 100000
#define N_EDGES 1600000

// ---------------- degree ----------------------------------------------------
__global__ void count_deg_kernel(const int* __restrict__ tgt, float* __restrict__ deg,
                                 int nEdges) {
  int e = blockIdx.x * blockDim.x + threadIdx.x;
  if (e < nEdges) atomicAdd(&deg[tgt[e]], 1.0f);
}

__global__ void invert_deg_kernel(float* deg, int n) {
  int i = blockIdx.x * blockDim.x + threadIdx.x;
  if (i < n) deg[i] = 1.0f / fmaxf(deg[i], 1.0f);
}

// ---------------- scatter-sum of 64-wide rows --------------------------------
// one lane per (edge, 4 features): float4 gather, 4 float atomics (L2 resident)
__global__ void scatter64_kernel(const float* __restrict__ x, const int* __restrict__ src,
                                 const int* __restrict__ tgt, float* __restrict__ agg,
                                 int nEdges) {
  long long i = (long long)blockIdx.x * blockDim.x + threadIdx.x;
  if (i >= (long long)nEdges * 16) return;
  int e = (int)(i >> 4);
  int f = (int)(i & 15) * 4;
  int s = src[e];
  int t = tgt[e];
  const float4 v = *(const float4*)(x + (size_t)s * 64 + f);
  float* dst = agg + (size_t)t * 64 + f;
  atomicAdd(dst + 0, v.x);
  atomicAdd(dst + 1, v.y);
  atomicAdd(dst + 2, v.z);
  atomicAdd(dst + 3, v.w);
}

// ---------------- weight pre-pack into B-fragment order ----------------------
// B (32x16 f16) order packed here (k order matches A loads): lane L holds column N=L%16, halves j -> row
// K = kb + (L>=16 ? 16 : 0) + j.  Packed element p = ((kb2*NT + t)*32 + L)*16 + j
// so each lane reads its 16 halves (32 B) contiguously.
template <int NT>
__global__ void pack_weight_kernel(const float* __restrict__ W, _Float16* __restrict__ out,
                                   int dimIn) {
  constexpr int DIM_OUT = NT * 16;
  int p = blockIdx.x * blockDim.x + threadIdx.x;
  int total = dimIn * DIM_OUT;
  if (p >= total) return;
  int j   = p & 15;
  int L   = (p >> 4) & 31;
  int t   = (p >> 9) % NT;
  int kb2 = p / (512 * NT);
  int K = kb2 * 32 + ((L >= 16) ? 16 : 0) + j;
  int N = t * 16 + (L & 15);
  out[p] = (_Float16)W[(size_t)K * DIM_OUT + N];
}

// ---------------- fused SAGE layer: out = l2norm(agg/deg@Wl + b + x@Wr) ------
// One wave computes a 32-node x (NT*16)-feature tile (2 WMMA M-tiles), so every
// B fragment loaded is consumed by two WMMAs.  DIM_IN fixed at 64 -> fully
// unrolled; all loads can be clause-batched ahead of the WMMA chain.
// A (16x32 f16) k order chosen here, same as B pack: lane L row M=L%16; halves j -> K=kb+(L>=16?16:0)+j.
// C/D: VGPR v lane L -> row v+(L>=16?8:0),
// column L%16.
template <int NT, bool RELU>
__global__ void sage_layer_kernel(const float* __restrict__ agg, const float* __restrict__ xin,
                                  const float* __restrict__ invdeg,
                                  const _Float16* __restrict__ pWl,
                                  const float* __restrict__ bias,
                                  const _Float16* __restrict__ pWr, float* __restrict__ xout,
                                  int nNodes) {
  constexpr int DIM_IN  = 64;
  constexpr int DIM_OUT = NT * 16;
  const int lane = threadIdx.x & 31;
  const int wid  = (blockIdx.x * blockDim.x + threadIdx.x) >> 5;
  const int m0   = wid * 32;           // two 16-row tiles per wave
  if (m0 >= nNodes) return;            // uniform whole-wave exit

  const int mrow  = lane & 15;         // A row / C column index
  const int khalf = (lane >> 4) << 4;  // A-frag K sub-offset: 0 or 16

  float scale[2];
  scale[0] = invdeg[m0 + mrow];
  scale[1] = invdeg[m0 + 16 + mrow];

  v8f acc[2][NT];
#pragma unroll
  for (int t = 0; t < NT; ++t) {
    float bv = bias[t * 16 + mrow];    // bias broadcast along rows
#pragma unroll
    for (int e = 0; e < 8; ++e) { acc[0][t][e] = bv; acc[1][t][e] = bv; }
  }

  const v16h* fragL = (const v16h*)pWl;
  const v16h* fragR = (const v16h*)pWr;

#pragma unroll
  for (int kb2 = 0; kb2 < 2; ++kb2) {
    const int kb = kb2 * 32;
    v16h aA[2], aX[2];
#pragma unroll
    for (int m = 0; m < 2; ++m) {
      const float* aggRow = agg + (size_t)(m0 + m * 16 + mrow) * DIM_IN;
      const float* xRow   = xin + (size_t)(m0 + m * 16 + mrow) * DIM_IN;
      float av[16], xv[16];
      *(float4*)(av + 0)  = *(const float4*)(aggRow + kb + khalf + 0);
      *(float4*)(av + 4)  = *(const float4*)(aggRow + kb + khalf + 4);
      *(float4*)(av + 8)  = *(const float4*)(aggRow + kb + khalf + 8);
      *(float4*)(av + 12) = *(const float4*)(aggRow + kb + khalf + 12);
      *(float4*)(xv + 0)  = *(const float4*)(xRow + kb + khalf + 0);
      *(float4*)(xv + 4)  = *(const float4*)(xRow + kb + khalf + 4);
      *(float4*)(xv + 8)  = *(const float4*)(xRow + kb + khalf + 8);
      *(float4*)(xv + 12) = *(const float4*)(xRow + kb + khalf + 12);
#pragma unroll
      for (int j = 0; j < 16; ++j) {
        aA[m][j] = (_Float16)(av[j] * scale[m]);
        aX[m][j] = (_Float16)xv[j];
      }
    }
#pragma unroll
    for (int t = 0; t < NT; ++t) {
      // pre-packed fragments: one contiguous 32 B vector per lane, reused 2x
      v16h bL = fragL[(size_t)(kb2 * NT + t) * 32 + lane];
      v16h bR = fragR[(size_t)(kb2 * NT + t) * 32 + lane];
#pragma unroll
      for (int m = 0; m < 2; ++m) {
        acc[m][t] = __builtin_amdgcn_wmma_f32_16x16x32_f16(false, aA[m], false, bL,
                                                           (short)0, acc[m][t], false, false);
        acc[m][t] = __builtin_amdgcn_wmma_f32_16x16x32_f16(false, aX[m], false, bR,
                                                           (short)0, acc[m][t], false, false);
      }
    }
  }

  // per-row L2 norm + store for each M tile
  const int rbase = (lane >> 4) << 3;  // rows 0..7 in lanes 0-15, 8..15 in 16-31
#pragma unroll
  for (int m = 0; m < 2; ++m) {
    float ss[8];
#pragma unroll
    for (int v = 0; v < 8; ++v) {
      float s = 0.0f;
#pragma unroll
      for (int t = 0; t < NT; ++t) s += acc[m][t][v] * acc[m][t][v];
      ss[v] = s;
    }
#pragma unroll
    for (int mask = 1; mask <= 8; mask <<= 1) {
#pragma unroll
      for (int v = 0; v < 8; ++v) ss[v] += __shfl_xor(ss[v], mask);
    }
#pragma unroll
    for (int v = 0; v < 8; ++v) {
      float inv = 1.0f / fmaxf(sqrtf(ss[v]), 1e-12f);
#pragma unroll
      for (int t = 0; t < NT; ++t) {
        float f = acc[m][t][v] * inv;
        if (RELU) f = fmaxf(f, 0.0f);
        xout[(size_t)(m0 + m * 16 + rbase + v) * DIM_OUT + t * 16 + mrow] = f;
      }
    }
  }
}

// ---------------- column sum of h3 [N,32] -----------------------------------
__global__ void colsum_kernel(const float* __restrict__ h, float* __restrict__ colsum, int n) {
  __shared__ float s[32];
  int tid = threadIdx.x;
  if (tid < 32) s[tid] = 0.0f;
  __syncthreads();
  int lane = tid & 31;
  int nw   = (gridDim.x * blockDim.x) >> 5;
  int wid  = (blockIdx.x * blockDim.x + tid) >> 5;
  float acc = 0.0f;
  for (int node = wid; node < n; node += nw) acc += h[(size_t)node * 32 + lane];
  atomicAdd(&s[lane], acc);
  __syncthreads();
  if (tid < 32) atomicAdd(&colsum[tid], s[tid]);
}

// context = tanh(mean(h) @ W_att)   (mean commutes with the linear map)
__global__ void context_kernel(const float* __restrict__ colsum,
                               const float* __restrict__ Watt, float* __restrict__ context) {
  int j = threadIdx.x;  // 32 threads
  float s = 0.0f;
#pragma unroll
  for (int k = 0; k < 32; ++k) s += colsum[k] * Watt[k * 32 + j];
  context[j] = tanhf(s * (1.0f / (float)N_NODES));
}

// pooled[j] = sum_i sigmoid(h_i . context) * h_i[j]
__global__ void attpool_kernel(const float* __restrict__ h, const float* __restrict__ context,
                               float* __restrict__ pooled, int n) {
  __shared__ float s[32];
  int tid = threadIdx.x;
  if (tid < 32) s[tid] = 0.0f;
  __syncthreads();
  int lane  = tid & 31;
  float ctx = context[lane];
  int nw    = (gridDim.x * blockDim.x) >> 5;
  int wid   = (blockIdx.x * blockDim.x + tid) >> 5;
  float acc = 0.0f;
  for (int node = wid; node < n; node += nw) {
    float v = h[(size_t)node * 32 + lane];
    float p = v * ctx;
#pragma unroll
    for (int mask = 16; mask >= 1; mask >>= 1) p += __shfl_xor(p, mask);
    float att = 1.0f / (1.0f + __expf(-p));
    acc += att * v;
  }
  atomicAdd(&s[lane], acc);
  __syncthreads();
  if (tid < 32) atomicAdd(&pooled[tid], s[tid]);
}

// score = sigmoid(relu(pooled@W_fc + b_fc) @ W_s + b_s)
__global__ void head_kernel(const float* __restrict__ pooled, const float* __restrict__ Wfc,
                            const float* __restrict__ bfc, const float* __restrict__ Ws,
                            const float* __restrict__ bs, float* __restrict__ out) {
  __shared__ float hid[16];
  int j = threadIdx.x;  // 32 threads
  if (j < 16) {
    float s = bfc[j];
#pragma unroll
    for (int k = 0; k < 32; ++k) s += pooled[k] * Wfc[k * 16 + j];
    hid[j] = fmaxf(s, 0.0f);
  }
  __syncthreads();
  if (j == 0) {
    float s = bs[0];
#pragma unroll
    for (int k = 0; k < 16; ++k) s += hid[k] * Ws[k];
    out[0] = 1.0f / (1.0f + expf(-s));
  }
}

// ---------------------------------------------------------------------------
extern "C" void kernel_launch(void* const* d_in, const int* in_sizes, int n_in,
                              void* d_out, int out_size, void* d_ws, size_t ws_size,
                              hipStream_t stream) {
  const float* feat = (const float*)d_in[0];
  const int*   eidx = (const int*)d_in[1];
  const float* W1l  = (const float*)d_in[2];
  const float* b1   = (const float*)d_in[3];
  const float* W1r  = (const float*)d_in[4];
  const float* W2l  = (const float*)d_in[5];
  const float* b2   = (const float*)d_in[6];
  const float* W2r  = (const float*)d_in[7];
  const float* W3l  = (const float*)d_in[8];
  const float* b3   = (const float*)d_in[9];
  const float* W3r  = (const float*)d_in[10];
  const float* Watt = (const float*)d_in[11];
  const float* Wfc  = (const float*)d_in[12];
  const float* bfc  = (const float*)d_in[13];
  const float* Ws   = (const float*)d_in[14];
  const float* bs   = (const float*)d_in[15];

  const int* src = eidx;            // edge_index[0]
  const int* tgt = eidx + N_EDGES;  // edge_index[1]

  // workspace carve (~90 MB + 40 KB packed weights)
  char* ws = (char*)d_ws;
  size_t off = 0;
  auto carve = [&](size_t bytes) -> char* {
    char* p = ws + off;
    off += (bytes + 255) & ~(size_t)255;
    return p;
  };
  float* agg    = (float*)carve((size_t)N_NODES * 64 * 4);
  float* invdeg = (float*)carve((size_t)N_NODES * 4);
  float* h1     = (float*)carve((size_t)N_NODES * 64 * 4);
  float* h2     = (float*)carve((size_t)N_NODES * 64 * 4);
  float* h3     = (float*)carve((size_t)N_NODES * 32 * 4);
  float* small  = (float*)carve(128 * 4);
  float* colsum  = small;
  float* context = small + 32;
  float* pooled  = small + 64;
  _Float16* pW1l = (_Float16*)carve(64 * 64 * 2);
  _Float16* pW1r = (_Float16*)carve(64 * 64 * 2);
  _Float16* pW2l = (_Float16*)carve(64 * 64 * 2);
  _Float16* pW2r = (_Float16*)carve(64 * 64 * 2);
  _Float16* pW3l = (_Float16*)carve(64 * 32 * 2);
  _Float16* pW3r = (_Float16*)carve(64 * 32 * 2);

  const int scatterBlocks = (int)(((long long)N_EDGES * 16 + 255) / 256);
  const int gemmBlocks    = (N_NODES / 32 + 7) / 8;  // 8 waves (32-row tiles) per block

  // pre-pack f16 weight fragments (cheap, deterministic each call)
  pack_weight_kernel<4><<<(64 * 64 + 255) / 256, 256, 0, stream>>>(W1l, pW1l, 64);
  pack_weight_kernel<4><<<(64 * 64 + 255) / 256, 256, 0, stream>>>(W1r, pW1r, 64);
  pack_weight_kernel<4><<<(64 * 64 + 255) / 256, 256, 0, stream>>>(W2l, pW2l, 64);
  pack_weight_kernel<4><<<(64 * 64 + 255) / 256, 256, 0, stream>>>(W2r, pW2r, 64);
  pack_weight_kernel<2><<<(64 * 32 + 255) / 256, 256, 0, stream>>>(W3l, pW3l, 64);
  pack_weight_kernel<2><<<(64 * 32 + 255) / 256, 256, 0, stream>>>(W3r, pW3r, 64);

  // degrees once (edge structure shared by all layers)
  hipMemsetAsync(invdeg, 0, (size_t)N_NODES * 4, stream);
  count_deg_kernel<<<(N_EDGES + 255) / 256, 256, 0, stream>>>(tgt, invdeg, N_EDGES);
  invert_deg_kernel<<<(N_NODES + 255) / 256, 256, 0, stream>>>(invdeg, N_NODES);

  // layer 1: 64 -> 64, ReLU
  hipMemsetAsync(agg, 0, (size_t)N_NODES * 64 * 4, stream);
  scatter64_kernel<<<scatterBlocks, 256, 0, stream>>>(feat, src, tgt, agg, N_EDGES);
  sage_layer_kernel<4, true><<<gemmBlocks, 256, 0, stream>>>(agg, feat, invdeg, pW1l, b1, pW1r,
                                                             h1, N_NODES);
  // layer 2: 64 -> 64, ReLU
  hipMemsetAsync(agg, 0, (size_t)N_NODES * 64 * 4, stream);
  scatter64_kernel<<<scatterBlocks, 256, 0, stream>>>(h1, src, tgt, agg, N_EDGES);
  sage_layer_kernel<4, true><<<gemmBlocks, 256, 0, stream>>>(agg, h1, invdeg, pW2l, b2, pW2r,
                                                             h2, N_NODES);
  // layer 3: 64 -> 32, no ReLU
  hipMemsetAsync(agg, 0, (size_t)N_NODES * 64 * 4, stream);
  scatter64_kernel<<<scatterBlocks, 256, 0, stream>>>(h2, src, tgt, agg, N_EDGES);
  sage_layer_kernel<2, false><<<gemmBlocks, 256, 0, stream>>>(agg, h2, invdeg, pW3l, b3, pW3r,
                                                              h3, N_NODES);

  // attention pooling + head
  hipMemsetAsync(small, 0, 128 * 4, stream);
  colsum_kernel<<<512, 256, 0, stream>>>(h3, colsum, N_NODES);
  context_kernel<<<1, 32, 0, stream>>>(colsum, Watt, context);
  attpool_kernel<<<512, 256, 0, stream>>>(h3, context, pooled, N_NODES);
  head_kernel<<<1, 32, 0, stream>>>(pooled, Wfc, bfc, Ws, bs, (float*)d_out);
}
